// ModulatedConv2d_64948495450684
// MI455X (gfx1250) — matrix operand order repaired and checked by reading, not compiled, not run on hardware
//
#include <hip/hip_runtime.h>
#include <hip/hip_bf16.h>

typedef __attribute__((ext_vector_type(16))) _Float16 v16h_;
typedef __attribute__((ext_vector_type(8)))  _Float16 v8h_;
typedef __attribute__((ext_vector_type(8)))  float    v8f_;
typedef __attribute__((ext_vector_type(4)))  int      v4i_;

#define BATCH      8
#define CIN        512
#define COUT       512
#define HW         4096
#define HP         66                      /* padded spatial edge (64 + halo) */
#define PROWS      (HP * HP)               /* 4356 padded spatial positions   */
#define LIN_SCALE  0.044194173824159216f   /* 1/sqrt(512)  */
#define CONV_SCALE 0.014731391274719739f   /* 1/sqrt(4608) */

#if defined(__HIP_DEVICE_COMPILE__) && \
    __has_builtin(__builtin_amdgcn_global_load_async_to_lds_b128) && \
    __has_builtin(__builtin_amdgcn_s_wait_asynccnt)
#define HAS_ASYNC 1
#else
#define HAS_ASYNC 0
#endif

// async global->LDS 16B copy (per lane); falls back to sync uint4 copy.
__device__ __forceinline__ void async_cp16(const _Float16* g, _Float16* l) {
#if HAS_ASYNC
    __builtin_amdgcn_global_load_async_to_lds_b128(
        (__attribute__((address_space(1))) v4i_*)(unsigned long long)(g),
        (__attribute__((address_space(3))) v4i_*)(l),
        0, 0);
#else
    *(uint4*)l = *(const uint4*)g;
#endif
}

#if HAS_ASYNC
#define WAIT_ASYNC(n) __builtin_amdgcn_s_wait_asynccnt(n)
#else
#define WAIT_ASYNC(n)
#endif

// ---------------------------------------------------------------------------
// Kernel 1: s[b][i] = style[b] . mod_weight[i] * LIN_SCALE + mod_bias[i]
// also writes the s5 output (second tuple element) at d_out tail.
// ---------------------------------------------------------------------------
__global__ void k_style_linear(const float* __restrict__ style,
                               const float* __restrict__ modw,
                               const float* __restrict__ modb,
                               float* __restrict__ s_buf,
                               float* __restrict__ s5_out) {
    int idx = blockIdx.x * blockDim.x + threadIdx.x;   // 0..4095
    int b = idx >> 9;
    int i = idx & 511;
    const float* st = style + (size_t)b * 512;
    const float* mw = modw  + (size_t)i * 512;
    float acc = 0.f;
    #pragma unroll 4
    for (int j = 0; j < 512; ++j) acc += st[j] * mw[j];
    float v = acc * LIN_SCALE + modb[i];
    s_buf[idx]  = v;
    s5_out[idx] = v;
}

// ---------------------------------------------------------------------------
// Kernel 2: scale[b][co] = CONV_SCALE * rsqrt(CONV_SCALE^2 *
//                 sum_{ci,kk} weight[co][ci][kk]^2 * s[b][ci]^2 + 1e-8)
// ---------------------------------------------------------------------------
__global__ void k_demod(const float* __restrict__ weight,
                        const float* __restrict__ s_buf,
                        float* __restrict__ scale_buf) {
    int idx = blockIdx.x * blockDim.x + threadIdx.x;   // 0..4095
    int b  = idx >> 9;
    int co = idx & 511;
    const float* wrow = weight + (size_t)co * (512 * 9);
    const float* sb   = s_buf  + (size_t)b * 512;
    float acc = 0.f;
    for (int ci = 0; ci < 512; ++ci) {
        float sv = sb[ci];
        const float* wp = wrow + ci * 9;
        float ws = 0.f;
        #pragma unroll
        for (int t = 0; t < 9; ++t) { float wv = wp[t]; ws += wv * wv; }
        acc += ws * sv * sv;
    }
    scale_buf[idx] = CONV_SCALE * rsqrtf(CONV_SCALE * CONV_SCALE * acc + 1e-8f);
}

// ---------------------------------------------------------------------------
// Kernel 3: wf16[kk][cout][cin] = (f16) weight[cout][cin][kk]
// ---------------------------------------------------------------------------
__global__ void k_wcvt(const float* __restrict__ weight,
                       _Float16* __restrict__ wf16) {
    int idx = blockIdx.x * blockDim.x + threadIdx.x;   // over 9*512*512
    int kk  = idx / (COUT * CIN);
    int rem = idx - kk * (COUT * CIN);                 // cout*512 + cin
    wf16[idx] = (_Float16)weight[(size_t)rem * 9 + kk];
}

// ---------------------------------------------------------------------------
// Kernel 4a: zero the padded x tensor (so halo rows/cols are 0).
// ---------------------------------------------------------------------------
__global__ void k_zero(uint4* __restrict__ p) {
    uint4 z; z.x = 0u; z.y = 0u; z.z = 0u; z.w = 0u;
    p[(size_t)blockIdx.x * blockDim.x + threadIdx.x] = z;
}

// ---------------------------------------------------------------------------
// Kernel 4b: xpad[b][(h+1)*66 + (w+1)][ci] = (f16)(x[b][ci][h*64+w] * s[b][ci])
// (LDS transpose NCHW -> padded NHWC-f16 with modulation folded in)
// ---------------------------------------------------------------------------
__global__ void k_xmod(const float* __restrict__ x,
                       const float* __restrict__ s_buf,
                       _Float16* __restrict__ xpad) {
    __shared__ float tile[32][33];
    int b   = blockIdx.z;
    int c0  = blockIdx.y * 32;
    int sp0 = blockIdx.x * 32;
    int tid = threadIdx.x;
    int tr  = tid >> 5;          // 0..7
    int tc  = tid & 31;          // 0..31
    const float* xb = x + ((size_t)b * CIN + c0) * HW + sp0;
    #pragma unroll
    for (int p = 0; p < 4; ++p) {
        int c = p * 8 + tr;
        tile[c][tc] = xb[(size_t)c * HW + tc];
    }
    __syncthreads();
    const float* sb = s_buf + (size_t)b * CIN + c0;
    _Float16* xo = xpad + (size_t)b * PROWS * CIN + c0;
    #pragma unroll
    for (int p = 0; p < 4; ++p) {
        int spr = p * 8 + tr;
        int sp  = sp0 + spr;
        int prow = ((sp >> 6) + 1) * HP + (sp & 63) + 1;
        xo[(size_t)prow * CIN + tc] = (_Float16)(tile[tc][spr] * sb[tc]);
    }
}

// ---------------------------------------------------------------------------
// Kernel 5: implicit-GEMM grouped conv via v_wmma_f32_16x16x32_f16 with
// double-buffered LDS staging (async global->LDS when available).
// Per batch: out[co][sp] = scale[b][co] * sum_{kk,ci} W*x  (halo-padded x)
// Block: 256 threads (8 wave32), tile BM=128 x BN=128, K-step 32.
// 144 K-stages total = 9 kernel taps x 16 cin-blocks.
// ---------------------------------------------------------------------------
#define LDA     40   /* halves per LDS row: 80B = 20 dwords, conflict-free */
#define NSTAGE  144

__global__ void __launch_bounds__(256)
k_conv_wmma(const _Float16* __restrict__ wf16,
            const _Float16* __restrict__ xpad,
            const float* __restrict__ scale_buf,
            float* __restrict__ out) {
    __shared__ _Float16 As[2][128 * LDA];   // [cout_row][k]
    __shared__ _Float16 Bs[2][128 * LDA];   // [sp_row  ][k]

    const int tid  = threadIdx.x;
    const int lane = tid & 31;
    const int wid  = tid >> 5;           // 0..7
    const int wm   = wid & 1;            // 2 waves along M (64 each)
    const int wn   = wid >> 1;           // 4 waves along N (32 each)
    const int l16  = lane & 15;
    const int hi   = (lane >> 4) & 1;    // lane half selects K-chunk pattern

    const int n0 = blockIdx.x * 128;     // spatial tile base
    const int m0 = blockIdx.y * 128;     // cout tile base
    const int b  = blockIdx.z;

    const _Float16* xb = xpad + (size_t)b * PROWS * CIN;

    // Stage-load indexing: 512 16B chunks per 8KB tile, 2 per thread.
    const int r_a = tid >> 2;            // row 0..63 (+64 on second chunk)
    const int c_a = (tid & 3) * 8;       // k-offset in halves

    // Padded-row index for this thread's two B rows (no masking needed).
    int pb[2];
    #pragma unroll
    for (int it = 0; it < 2; ++it) {
        int sp = n0 + r_a + it * 64;
        pb[it] = ((sp >> 6) + 1) * HP + (sp & 63) + 1;
    }

    v8f_ acc[4][2];
    #pragma unroll
    for (int tm = 0; tm < 4; ++tm)
        #pragma unroll
        for (int tn = 0; tn < 2; ++tn)
            acc[tm][tn] = (v8f_){0.f,0.f,0.f,0.f,0.f,0.f,0.f,0.f};

    // Issue one K-stage's global->LDS staging (4 x b128 per thread).
    auto issue_stage = [&](int s, int bufIdx) {
        int kk = s >> 4;                   // kernel tap 0..8
        int k0 = (s & 15) << 5;            // cin block 0..480
        int dy = kk / 3 - 1, dx = kk % 3 - 1;
        int shiftp = dy * HP + dx;
        const _Float16* gA = wf16 + (size_t)kk * (COUT * CIN)
                                  + (size_t)m0 * CIN + k0 + c_a;
        const _Float16* gB = xb + k0 + c_a;
        _Float16* lA = &As[bufIdx][c_a];
        _Float16* lB = &Bs[bufIdx][c_a];
        #pragma unroll
        for (int it = 0; it < 2; ++it) {
            int r = r_a + it * 64;
            async_cp16(gA + (size_t)r * CIN,                 &lA[r * LDA]);
            async_cp16(gB + (size_t)(pb[it] + shiftp) * CIN, &lB[r * LDA]);
        }
    };

    issue_stage(0, 0);                     // prologue

    for (int s = 0; s < NSTAGE; ++s) {
        const int cur = s & 1;
        if (s < NSTAGE - 1) {
            issue_stage(s + 1, cur ^ 1);
            WAIT_ASYNC(4);                 // stage s landed; s+1 in flight
        } else {
            WAIT_ASYNC(0);
        }
        __syncthreads();

        // ---- fragment loads per ISA 7.12.2 16-bit layouts ----
        v16h_ afr[4];
        #pragma unroll
        for (int tm = 0; tm < 4; ++tm) {
            const _Float16* ap = &As[cur][(wm * 64 + tm * 16 + l16) * LDA + (hi ? 8 : 0)];
            v8h_ alo = *(const v8h_*)(ap);        // K[0..7]   / K[8..15]
            v8h_ ahi = *(const v8h_*)(ap + 16);   // K[16..23] / K[24..31]
            afr[tm] = __builtin_shufflevector(alo, ahi,
                      0,1,2,3,4,5,6,7,8,9,10,11,12,13,14,15);
        }
        v16h_ bfr[2];
        #pragma unroll
        for (int tn = 0; tn < 2; ++tn) {
            const _Float16* bp = &Bs[cur][(wn * 32 + tn * 16 + l16) * LDA + hi * 8];
            v8h_ blo = *(const v8h_*)(bp);        // K[0..7]   / K[8..15]
            v8h_ bhi = *(const v8h_*)(bp + 16);   // K[16..23] / K[24..31]
            bfr[tn] = __builtin_shufflevector(blo, bhi,
                      0,1,2,3,4,5,6,7,8,9,10,11,12,13,14,15);
        }

        // ---- 8 WMMAs: 64x32 wave tile, f32 accumulate ----
        #pragma unroll
        for (int tm = 0; tm < 4; ++tm)
            #pragma unroll
            for (int tn = 0; tn < 2; ++tn)
                acc[tm][tn] = __builtin_amdgcn_wmma_f32_16x16x32_f16(
                    false, afr[tm], false, bfr[tn],
                    (short)0, acc[tm][tn], false, false);

        __syncthreads();                   // buf[cur] free for stage s+2
    }

    // ---- epilogue: apply conv_scale*demod, store f32 ----
    const float* scl = scale_buf + (size_t)b * COUT;
    float* outp = out + (size_t)b * COUT * HW;
    #pragma unroll
    for (int tm = 0; tm < 4; ++tm) {
        #pragma unroll
        for (int tn = 0; tn < 2; ++tn) {
            int mbase = m0 + wm * 64 + tm * 16 + hi * 8;
            int n     = n0 + wn * 32 + tn * 16 + l16;
            #pragma unroll
            for (int i = 0; i < 8; ++i) {
                int m = mbase + i;
                outp[(size_t)m * HW + n] = acc[tm][tn][i] * scl[m];
            }
        }
    }
}

// ---------------------------------------------------------------------------
extern "C" void kernel_launch(void* const* d_in, const int* in_sizes, int n_in,
                              void* d_out, int out_size, void* d_ws, size_t ws_size,
                              hipStream_t stream) {
    const float* x      = (const float*)d_in[0];
    const float* style  = (const float*)d_in[1];
    const float* weight = (const float*)d_in[2];
    const float* modw   = (const float*)d_in[3];
    const float* modb   = (const float*)d_in[4];
    float* out = (float*)d_out;

    char* ws = (char*)d_ws;
    float*    s_buf     = (float*)(ws);                       // 16 KB
    float*    scale_buf = (float*)(ws + 16384);               // 16 KB
    _Float16* wf16      = (_Float16*)(ws + 32768);            // 4.5 MB
    _Float16* xpad      = (_Float16*)(ws + 4751360);          // 8*4356*512*2 = 34 MB

    float* s5_out = out + (size_t)BATCH * COUT * HW;          // second tuple output

    k_style_linear<<<16, 256, 0, stream>>>(style, modw, modb, s_buf, s5_out);
    k_demod<<<16, 256, 0, stream>>>(weight, s_buf, scale_buf);
    k_wcvt<<<(9 * COUT * CIN) / 256, 256, 0, stream>>>(weight, wf16);
    // zero padded x (8*4356*512*2 bytes = 2,230,272 uint4 chunks)
    k_zero<<<(BATCH * PROWS * CIN * 2 / 16) / 256, 256, 0, stream>>>((uint4*)xpad);
    dim3 gx(HW / 32, CIN / 32, BATCH);
    k_xmod<<<gx, 256, 0, stream>>>(x, s_buf, xpad);
    dim3 gc(HW / 128, COUT / 128, BATCH);
    k_conv_wmma<<<gc, 256, 0, stream>>>(wf16, xpad, scale_buf, out);
}
